// InformerStandard_31997506355458
// MI455X (gfx1250) — matrix operand order repaired and checked by reading, not compiled, not run on hardware
//
#include <hip/hip_runtime.h>
#include <hip/hip_bf16.h>
#include <math.h>

// Problem constants (from reference)
#define BB 8
#define SS 1024   // sequence length after the reference's transpose
#define DM 1024   // model dim
#define HH 16
#define DK 64
#define UU 6      // int(ln(1024)) = 6
#define FFD 32
#define NL 2
#define HZ 96
#define PP 16

typedef __attribute__((ext_vector_type(16))) __bf16 v16bf;
typedef __attribute__((ext_vector_type(8)))  float  v8f;
typedef __attribute__((ext_vector_type(4)))  unsigned int u32x4;

union FragBF { v16bf v; u32x4 q[2]; };

__device__ __forceinline__ unsigned short f2bf(float x){
  union { float f; unsigned int u; } c; c.f = x;
  unsigned int u = c.u;
  u += 0x7fffu + ((u >> 16) & 1u);   // round-to-nearest-even
  return (unsigned short)(u >> 16);
}
__device__ __forceinline__ float bf2f(unsigned short h){
  union { unsigned int u; float f; } c; c.u = ((unsigned int)h) << 16;
  return c.f;
}

// ---------------- embedding + positional encoding (transposed per reference) -------------
// h[b,i,j] = sum_p x[b,p,j]*emb_W[p,i] + emb_b[i] + pe[i,j]
__global__ void k_embed(const float* __restrict__ x, const float* __restrict__ embW,
                        const float* __restrict__ embB,
                        float* __restrict__ h, unsigned short* __restrict__ hbf){
  int row = blockIdx.x;                  // b*SS + i
  int b = row >> 10; int i = row & 1023;
  __shared__ float ew[PP];
  if (threadIdx.x < PP) ew[threadIdx.x] = embW[threadIdx.x * DM + i];
  __syncthreads();
  float eb = embB[i];
  const float kln = 9.2103403719761836f / 1024.0f;   // ln(10000)/1024
  for (int j = threadIdx.x; j < DM; j += blockDim.x){
    float acc = eb;
    #pragma unroll
    for (int p = 0; p < PP; p++) acc += ew[p] * x[(b*PP + p)*SS + j];
    int k2 = j & ~1;
    float div = expf(-(float)k2 * kln);
    float ang = (float)i * div;
    acc += (j & 1) ? cosf(ang) : sinf(ang);
    h[row*DM + j]   = acc;
    hbf[row*DM + j] = f2bf(acc);
  }
}

// ---------------- pack f32 weight (K x N row-major) into WMMA B-fragment order -----------
// packed index: (((kstep*ntiles + ntile)*32 + lane)*16 + e)
//   k = kstep*32 + (lane>>4)*16 + e ; n = ntile*16 + (lane&15)
__global__ void k_pack_b(const float* __restrict__ W, unsigned short* __restrict__ P,
                         int K, int N){
  int idx = blockIdx.x * blockDim.x + threadIdx.x;
  if (idx >= K * N) return;
  int e    = idx & 15;
  int lane = (idx >> 4) & 31;
  int rest = idx >> 9;
  int ntiles = N >> 4;
  int ntile = rest % ntiles;
  int kstep = rest / ntiles;
  int k = kstep*32 + (lane >> 4)*16 + e;
  int n = ntile*16 + (lane & 15);
  P[idx] = f2bf(W[k*N + n]);
}

// ---------------- register-blocked WMMA GEMM ---------------------------------------------
// out = act(A_bf16[M,K] @ Bpacked + bias). 128 threads = 4 waves.
// Each wave computes a 16 x (16*NBLK) output strip: one A fragment feeds NBLK WMMAs/k-step.
template <int NBLK>
__global__ void k_gemm(const unsigned short* __restrict__ A,
                       const unsigned short* __restrict__ Bp,
                       const float* __restrict__ bias,
                       unsigned short* __restrict__ outBf,
                       float* __restrict__ outF,
                       int M, int N, int K, int relu){
  int wave = threadIdx.x >> 5;
  int lane = threadIdx.x & 31;
  int l16 = lane & 15, half = lane >> 4;
  int ntiles = N >> 4;
  int nsuper = ntiles / NBLK;
  int tile = blockIdx.x * 4 + wave;
  int mt = tile / nsuper, ntb = tile % nsuper;
  if (mt >= (M >> 4)) return;
  const unsigned short* Arow = A + (size_t)(mt*16 + l16) * K;
  v8f acc[NBLK];
  #pragma unroll
  for (int j = 0; j < NBLK; j++) acc[j] = (v8f){};
  int ksteps = K >> 5;
  for (int ks = 0; ks < ksteps; ks++){
    int k0 = ks << 5;
    FragBF a;
    // A fragment in this code's k order (same as packed B): lanes hold rows l16; K runs [half*16, half*16+15]
    a.q[0] = *(const u32x4*)(Arow + k0 + half*16);
    a.q[1] = *(const u32x4*)(Arow + k0 + half*16 + 8);
    const unsigned short* bp0 = Bp + ((size_t)(ks*ntiles + ntb*NBLK)*32 + lane)*16;
    #pragma unroll
    for (int j = 0; j < NBLK; j++){
      FragBF b;
      b.q[0] = *(const u32x4*)(bp0 + (size_t)j*512);
      b.q[1] = *(const u32x4*)(bp0 + (size_t)j*512 + 8);
      acc[j] = __builtin_amdgcn_wmma_f32_16x16x32_bf16(false, a.v, false, b.v,
                                                       (short)0, acc[j], false, false);
    }
  }
  #pragma unroll
  for (int j = 0; j < NBLK; j++){
    int col = (ntb*NBLK + j)*16 + l16;
    float bvv = bias ? bias[col] : 0.0f;
    #pragma unroll
    for (int r = 0; r < 8; r++){
      int rowm = mt*16 + half*8 + r;
      float v = acc[j][r] + bvv;
      if (relu) v = fmaxf(v, 0.0f);
      if (outBf) outBf[(size_t)rowm*N + col] = f2bf(v);
      if (outF)  outF [(size_t)rowm*N + col] = v;
    }
  }
}

// ---------------- streaming row-max of scaled QK^T (no score materialization) ------------
// grid = B*H*(SS/16), 32 threads (1 wave). WMMA per 16x16 score tile, running max.
__global__ void k_rowmax(const unsigned short* __restrict__ Q,
                         const unsigned short* __restrict__ Kb,
                         float* __restrict__ Mout){
  int st = blockIdx.x & 63;
  int hh = (blockIdx.x >> 6) & 15;
  int b  = blockIdx.x >> 10;
  int lane = threadIdx.x & 31;
  int l16 = lane & 15, half = lane >> 4;
  const unsigned short* Qb  = Q  + (size_t)(b*SS + st*16)*DM + hh*DK;
  const unsigned short* Kbb = Kb + (size_t)b*SS*DM + hh*DK;
  float rmax[8];
  #pragma unroll
  for (int r = 0; r < 8; r++) rmax[r] = -3.0e38f;
  const unsigned short* Aq = Qb + l16*DM;
  FragBF a0, a1;
  a0.q[0] = *(const u32x4*)(Aq + half*16);
  a0.q[1] = *(const u32x4*)(Aq + half*16 + 8);
  a1.q[0] = *(const u32x4*)(Aq + 32 + half*16);
  a1.q[1] = *(const u32x4*)(Aq + 32 + half*16 + 8);
  for (int mtile = 0; mtile < 64; mtile++){
    const unsigned short* Kr = Kbb + (size_t)(mtile*16 + l16)*DM;
    FragBF b0, b1;
    b0.q[0] = *(const u32x4*)(Kr + half*16);
    b0.q[1] = *(const u32x4*)(Kr + half*16 + 8);
    b1.q[0] = *(const u32x4*)(Kr + 32 + half*16);
    b1.q[1] = *(const u32x4*)(Kr + 32 + half*16 + 8);
    v8f acc = {};
    acc = __builtin_amdgcn_wmma_f32_16x16x32_bf16(false, a0.v, false, b0.v,
                                                  (short)0, acc, false, false);
    acc = __builtin_amdgcn_wmma_f32_16x16x32_bf16(false, a1.v, false, b1.v,
                                                  (short)0, acc, false, false);
    #pragma unroll
    for (int r = 0; r < 8; r++) rmax[r] = fmaxf(rmax[r], acc[r]);
  }
  #pragma unroll
  for (int off = 1; off < 16; off <<= 1){
    #pragma unroll
    for (int r = 0; r < 8; r++)
      rmax[r] = fmaxf(rmax[r], __shfl_xor(rmax[r], off, 16));
  }
  if (l16 == 0){
    #pragma unroll
    for (int r = 0; r < 8; r++)
      Mout[((b*HH + hh)*SS) + st*16 + half*8 + r] = rmax[r] * 0.125f;
  }
}

// ---------------- top-6 per (b,h): sequential argmax (matches jax top_k tie order) -------
__global__ void k_top6(const float* __restrict__ Mv, int* __restrict__ top){
  int bh = blockIdx.x * blockDim.x + threadIdx.x;
  if (bh >= BB*HH) return;
  const float* m = Mv + bh*SS;
  int sel[UU];
  for (int j = 0; j < UU; j++){
    float best = -3.4e38f; int bi = 0;
    for (int s = 0; s < SS; s++){
      bool taken = false;
      for (int t = 0; t < j; t++) if (sel[t] == s) taken = true;
      if (taken) continue;
      float v = m[s];
      if (v > best){ best = v; bi = s; }
    }
    sel[j] = bi; top[bh*UU + j] = bi;
  }
}

// ---------------- sparse attention on the 6 selected rows per head -----------------------
__global__ void k_spattn(const unsigned short* __restrict__ Q,
                         const unsigned short* __restrict__ Kb,
                         const unsigned short* __restrict__ V,
                         const int* __restrict__ top,
                         float* __restrict__ ctx){
  int id = blockIdx.x;                       // b*HH*UU + h*UU + u
  int hh = (id/UU) % HH; int b = id/(UU*HH);
  int s = top[id];
  __shared__ float qs[DK];
  __shared__ float sc[SS];
  __shared__ float red[128];
  int tid = threadIdx.x;
  if (tid < DK) qs[tid] = bf2f(Q[(size_t)(b*SS + s)*DM + hh*DK + tid]);
  __syncthreads();
  float lmax = -3.4e38f;
  for (int m = tid; m < SS; m += 128){
    const unsigned short* kr = Kb + (size_t)(b*SS + m)*DM + hh*DK;
    float acc = 0.f;
    #pragma unroll 8
    for (int k = 0; k < DK; k++) acc += qs[k]*bf2f(kr[k]);
    acc *= 0.125f;
    sc[m] = acc;
    lmax = fmaxf(lmax, acc);
  }
  red[tid] = lmax; __syncthreads();
  for (int o = 64; o > 0; o >>= 1){ if (tid < o) red[tid] = fmaxf(red[tid], red[tid+o]); __syncthreads(); }
  float mx = red[0]; __syncthreads();
  float lsum = 0.f;
  for (int m = tid; m < SS; m += 128){
    float e = expf(sc[m] - mx);
    sc[m] = e; lsum += e;
  }
  red[tid] = lsum; __syncthreads();
  for (int o = 64; o > 0; o >>= 1){ if (tid < o) red[tid] += red[tid+o]; __syncthreads(); }
  float inv = 1.0f / red[0];
  if (tid < DK){
    float acc = 0.f;
    for (int m = 0; m < SS; m++) acc += sc[m]*bf2f(V[(size_t)(b*SS + m)*DM + hh*DK + tid]);
    ctx[id*DK + tid] = acc * inv;
  }
}

// ---------------- y[row] = ctx[row] @ Wo[head block]  (64 -> 1024 matvec) ----------------
__global__ void k_ctxproj(const float* __restrict__ ctx, const float* __restrict__ Wo,
                          float* __restrict__ y){
  int id = blockIdx.x;
  int hh = (id/UU) % HH;
  __shared__ float cs[DK];
  if (threadIdx.x < DK) cs[threadIdx.x] = ctx[id*DK + threadIdx.x];
  __syncthreads();
  for (int f = threadIdx.x; f < DM; f += blockDim.x){
    float acc = 0.f;
    #pragma unroll 8
    for (int k = 0; k < DK; k++) acc += cs[k]*Wo[(size_t)(hh*DK + k)*DM + f];
    y[(size_t)id*DM + f] = acc;
  }
}

__global__ void k_fillbias(const float* __restrict__ b, float* __restrict__ out){
  size_t row = blockIdx.x;
  for (int j = threadIdx.x; j < DM; j += blockDim.x) out[row*DM + j] = b[j];
}

__global__ void k_scatter(const float* __restrict__ y, const int* __restrict__ top,
                          float* __restrict__ out){
  int id = blockIdx.x;
  int b = id/(UU*HH);
  int s = top[id];
  float* dst = out + (size_t)(b*SS + s)*DM;
  const float* src = y + (size_t)id*DM;
  for (int j = threadIdx.x; j < DM; j += blockDim.x) atomicAdd(dst + j, src[j]);
}

// ---------------- residual add + LayerNorm (biased var, eps=1e-5), refresh bf16 copy -----
__global__ void k_addln(float* __restrict__ h, const float* __restrict__ add,
                        const float* __restrict__ g, const float* __restrict__ beta,
                        unsigned short* __restrict__ hbf){
  size_t row = blockIdx.x;
  __shared__ float red[256];
  float v[4];
  int tid = threadIdx.x;
  float s = 0.f;
  #pragma unroll
  for (int i = 0; i < 4; i++){
    int j = tid + i*256;
    v[i] = h[row*DM + j] + add[row*DM + j];
    s += v[i];
  }
  red[tid] = s; __syncthreads();
  for (int o = 128; o > 0; o >>= 1){ if (tid < o) red[tid] += red[tid+o]; __syncthreads(); }
  float mean = red[0] * (1.0f/DM); __syncthreads();
  float s2 = 0.f;
  #pragma unroll
  for (int i = 0; i < 4; i++){ float d = v[i] - mean; s2 += d*d; }
  red[tid] = s2; __syncthreads();
  for (int o = 128; o > 0; o >>= 1){ if (tid < o) red[tid] += red[tid+o]; __syncthreads(); }
  float rstd = rsqrtf(red[0] * (1.0f/DM) + 1e-5f);
  #pragma unroll
  for (int i = 0; i < 4; i++){
    int j = tid + i*256;
    float o = (v[i] - mean)*rstd*g[j] + beta[j];
    h[row*DM + j]   = o;
    hbf[row*DM + j] = f2bf(o);
  }
}

__global__ void k_mean(const float* __restrict__ h, float* __restrict__ hm){
  int idx = blockIdx.x*blockDim.x + threadIdx.x;
  if (idx >= BB*DM) return;
  int b = idx / DM, f = idx % DM;
  float s = 0.f;
  for (int t = 0; t < SS; t++) s += h[(size_t)(b*SS + t)*DM + f];
  hm[idx] = s * (1.0f/SS);
}

__global__ void k_headout(const float* __restrict__ hm, const float* __restrict__ W,
                          const float* __restrict__ bo, float* __restrict__ out){
  int b = blockIdx.x; int t = threadIdx.x;
  if (t >= HZ) return;
  float acc = bo[t];
  for (int f = 0; f < DM; f++) acc += hm[b*DM + f]*W[f*HZ + t];
  out[b*HZ + t] = acc;
}

extern "C" void kernel_launch(void* const* d_in, const int* in_sizes, int n_in,
                              void* d_out, int out_size, void* d_ws, size_t ws_size,
                              hipStream_t stream) {
  const float* x    = (const float*)d_in[0];
  const float* embW = (const float*)d_in[1];
  const float* embB = (const float*)d_in[2];
  const float* Wq = (const float*)d_in[3];  const float* bq = (const float*)d_in[4];
  const float* Wk = (const float*)d_in[5];  const float* bk = (const float*)d_in[6];
  const float* Wv = (const float*)d_in[7];  const float* bv = (const float*)d_in[8];
  const float* Wo = (const float*)d_in[9];  const float* bo = (const float*)d_in[10];
  const float* W1 = (const float*)d_in[11]; const float* b1 = (const float*)d_in[12];
  const float* W2 = (const float*)d_in[13]; const float* b2 = (const float*)d_in[14];
  const float* ln1g = (const float*)d_in[15]; const float* ln1b = (const float*)d_in[16];
  const float* ln2g = (const float*)d_in[17]; const float* ln2b = (const float*)d_in[18];
  const float* outW = (const float*)d_in[19]; const float* outb = (const float*)d_in[20];

  // workspace carve
  char* p = (char*)d_ws;
  auto alloc = [&](size_t bytes) -> void* {
    void* r = (void*)p; p += (bytes + 255) & ~(size_t)255; return r;
  };
  float*          h     = (float*)         alloc((size_t)BB*SS*DM*4);
  unsigned short* hbf   = (unsigned short*)alloc((size_t)BB*SS*DM*2);
  unsigned short* Qbf   = (unsigned short*)alloc((size_t)BB*SS*DM*2);
  unsigned short* Kbf   = (unsigned short*)alloc((size_t)BB*SS*DM*2);
  unsigned short* Vbf   = (unsigned short*)alloc((size_t)BB*SS*DM*2);
  float*          abuf  = (float*)         alloc((size_t)BB*SS*DM*4);
  unsigned short* midbf = (unsigned short*)alloc((size_t)BB*SS*FFD*2);
  float*          rmax  = (float*)         alloc((size_t)BB*HH*SS*4);
  int*            top   = (int*)           alloc((size_t)BB*HH*UU*4);
  float*          ctx   = (float*)         alloc((size_t)BB*HH*UU*DK*4);
  float*          ybuf  = (float*)         alloc((size_t)BB*HH*UU*DM*4);
  float*          hm    = (float*)         alloc((size_t)BB*DM*4);
  unsigned short* pq    = (unsigned short*)alloc((size_t)NL*DM*DM*2);
  unsigned short* pk    = (unsigned short*)alloc((size_t)NL*DM*DM*2);
  unsigned short* pv    = (unsigned short*)alloc((size_t)NL*DM*DM*2);
  unsigned short* p1    = (unsigned short*)alloc((size_t)NL*DM*FFD*2);
  unsigned short* p2    = (unsigned short*)alloc((size_t)NL*FFD*DM*2);
  (void)in_sizes; (void)n_in; (void)out_size; (void)ws_size;

  // pack all layer weights to WMMA bf16 fragment order
  for (int l = 0; l < NL; l++){
    k_pack_b<<<(DM*DM + 255)/256, 256, 0, stream>>>(Wq + (size_t)l*DM*DM, pq + (size_t)l*DM*DM, DM, DM);
    k_pack_b<<<(DM*DM + 255)/256, 256, 0, stream>>>(Wk + (size_t)l*DM*DM, pk + (size_t)l*DM*DM, DM, DM);
    k_pack_b<<<(DM*DM + 255)/256, 256, 0, stream>>>(Wv + (size_t)l*DM*DM, pv + (size_t)l*DM*DM, DM, DM);
    k_pack_b<<<(DM*FFD + 255)/256, 256, 0, stream>>>(W1 + (size_t)l*DM*FFD, p1 + (size_t)l*DM*FFD, DM, FFD);
    k_pack_b<<<(FFD*DM + 255)/256, 256, 0, stream>>>(W2 + (size_t)l*FFD*DM, p2 + (size_t)l*FFD*DM, FFD, DM);
  }

  k_embed<<<BB*SS, 256, 0, stream>>>(x, embW, embB, h, hbf);

  const int M = BB*SS;
  // NBLK=4 strips: tiles = (M/16) * (N/64); 4 waves per block
  const int gBig = (M/16)*(DM/64)/4;    // 2048 blocks
  const int gFF1 = (M/16)*(FFD/32)/4;   // 128 blocks (NBLK=2)

  for (int l = 0; l < NL; l++){
    // Q, K, V projections (WMMA bf16, register-blocked 16x64 per wave)
    k_gemm<4><<<gBig, 128, 0, stream>>>(hbf, pq + (size_t)l*DM*DM, bq + l*DM, Qbf, nullptr, M, DM, DM, 0);
    k_gemm<4><<<gBig, 128, 0, stream>>>(hbf, pk + (size_t)l*DM*DM, bk + l*DM, Kbf, nullptr, M, DM, DM, 0);
    k_gemm<4><<<gBig, 128, 0, stream>>>(hbf, pv + (size_t)l*DM*DM, bv + l*DM, Vbf, nullptr, M, DM, DM, 0);
    // streaming row-max of scaled QK^T (WMMA), then top-6
    k_rowmax<<<BB*HH*(SS/16), 32, 0, stream>>>(Qbf, Kbf, rmax);
    k_top6<<<1, 128, 0, stream>>>(rmax, top);
    // sparse attention on selected rows; project through Wo; assemble output rows
    k_spattn<<<BB*HH*UU, 128, 0, stream>>>(Qbf, Kbf, Vbf, top, ctx);
    k_ctxproj<<<BB*HH*UU, 256, 0, stream>>>(ctx, Wo + (size_t)l*DM*DM, ybuf);
    k_fillbias<<<BB*SS, 256, 0, stream>>>(bo + l*DM, abuf);
    k_scatter<<<BB*HH*UU, 256, 0, stream>>>(ybuf, top, abuf);
    k_addln<<<BB*SS, 256, 0, stream>>>(h, abuf, ln1g + l*DM, ln1b + l*DM, hbf);
    // FFN: relu(h@W1+b1)@W2+b2, residual + LN2
    k_gemm<2><<<gFF1, 128, 0, stream>>>(hbf, p1 + (size_t)l*DM*FFD, b1 + l*FFD, midbf, nullptr, M, FFD, DM, 1);
    k_gemm<4><<<gBig, 128, 0, stream>>>(midbf, p2 + (size_t)l*FFD*DM, b2 + l*DM, nullptr, abuf, M, DM, FFD, 0);
    k_addln<<<BB*SS, 256, 0, stream>>>(h, abuf, ln2g + l*DM, ln2b + l*DM, hbf);
  }

  k_mean<<<(BB*DM + 255)/256, 256, 0, stream>>>(h, hm);
  k_headout<<<BB, 96, 0, stream>>>(hm, outW, outb, (float*)d_out);
}
